// EPMoE_84104049590645
// MI455X (gfx1250) — matrix operand order repaired and checked by reading, not compiled, not run on hardware
//
#include <hip/hip_runtime.h>
#include <hip/hip_bf16.h>

typedef __attribute__((ext_vector_type(16))) __bf16 v16bf;
typedef __attribute__((ext_vector_type(8)))  __bf16 v8bf;
typedef __attribute__((ext_vector_type(8)))  float  v8f;

#define EXPERTS   8
#define TOPK      2
#define HID       1024
#define FFN       2048
#define TOKENS    2048
#define NROWS     (TOKENS * TOPK)              /* 4096 expanded rows        */
#define TILE_M    64
#define MAX_TILES (NROWS / TILE_M + EXPERTS)   /* 72                        */
#define MAX_PROWS (NROWS + EXPERTS * TILE_M)   /* 4608 padded rows (>=4600) */

/* ------------------------------------------------------------------ */
/* WMMA helpers                                                        */
/* ------------------------------------------------------------------ */
__device__ __forceinline__ v8f wmma_bf16(v16bf a, v16bf b, v8f c) {
    // D = A(16x32 bf16) * B(32x16 bf16) + C(16x16 f32)
    return __builtin_amdgcn_wmma_f32_16x16x32_bf16(
        /*neg_a=*/false, a, /*neg_b=*/false, b,
        /*c_mod=*/(short)0, c, /*reuse_a=*/false, /*reuse_b=*/false);
}

// A fragment (16x32 bf16, row-major source): lane half selects K-half.
// elems 0..15 = K[kbase + hi*16 .. +15]
__device__ __forceinline__ v16bf load_a_frag(const __bf16* p) {
    v8bf lo = *(const v8bf*)p;
    v8bf hi = *(const v8bf*)(p + 8);
    return __builtin_shufflevector(lo, hi, 0,1,2,3,4,5,6,7,8,9,10,11,12,13,14,15);
}

// fast SiLU(g) * u : sigmoid via v_exp_f32 + v_rcp_f32 (no IEEE divide chain)
__device__ __forceinline__ float silu_mul(float g, float u) {
    const float s = __builtin_amdgcn_rcpf(1.0f + __expf(-g));
    return g * s * u;
}

/* ------------------------------------------------------------------ */
/* 1) fp32 -> bf16 transpose-convert: in[E][R][C] -> out[E][C][R]      */
/* ------------------------------------------------------------------ */
__global__ void __launch_bounds__(256)
cvt_transpose(const float* __restrict__ in, __bf16* __restrict__ out,
              int R, int C)
{
    __shared__ float tile[32][33];
    const size_t plane = (size_t)R * C;
    const float*  ip = in  + (size_t)blockIdx.z * plane;
    __bf16*       op = out + (size_t)blockIdx.z * plane;
    const int c0 = blockIdx.x * 32;
    const int r0 = blockIdx.y * 32;
    for (int j = threadIdx.y; j < 32; j += 8)
        tile[j][threadIdx.x] = ip[(size_t)(r0 + j) * C + c0 + threadIdx.x];
    __syncthreads();
    for (int j = threadIdx.y; j < 32; j += 8)
        op[(size_t)(c0 + j) * R + r0 + threadIdx.x] = (__bf16)tile[threadIdx.x][j];
}

/* ------------------------------------------------------------------ */
/* 2) routing prep: histogram, padded offsets, tile table, stable sort */
/*    meta[0]=numTiles meta[1]=totalPaddedRows meta[16+t]=tileExpert   */
/*    meta[96+t]=tileRowBase                                           */
/* ------------------------------------------------------------------ */
__global__ void __launch_bounds__(256)
moe_prep(const int* __restrict__ ids, int* __restrict__ meta,
         int* __restrict__ perm, int* __restrict__ inv)
{
    __shared__ int cnt[EXPERTS];
    __shared__ int base[EXPERTS];
    const int t = threadIdx.x;
    if (t < EXPERTS) cnt[t] = 0;
    __syncthreads();
    for (int i = t; i < NROWS; i += 256)
        atomicAdd(&cnt[ids[i] & (EXPERTS - 1)], 1);
    __syncthreads();
    if (t == 0) {
        int run = 0, nt = 0;
        for (int e = 0; e < EXPERTS; ++e) {
            base[e] = run;
            const int c = cnt[e];
            const int tiles = (c + TILE_M - 1) / TILE_M;
            for (int tt = 0; tt < tiles; ++tt) {
                meta[16 + nt] = e;
                meta[96 + nt] = run + tt * TILE_M;
                ++nt;
            }
            run += tiles * TILE_M;
        }
        meta[0] = nt;
        meta[1] = run;
    }
    __syncthreads();
    if (t < EXPERTS) {                       // one thread per expert: stable scatter
        const int e = t;
        int r = base[e];
        for (int i = 0; i < NROWS; ++i) {
            if ((ids[i] & (EXPERTS - 1)) == e) {
                perm[r] = i;                 // padded row -> flat (token*K+slot)
                inv[i]  = r;                 // flat -> padded row
                ++r;
            }
        }
        const int end = base[e] + ((cnt[e] + TILE_M - 1) / TILE_M) * TILE_M;
        for (; r < end; ++r) perm[r] = -1;   // pad rows
    }
}

/* ------------------------------------------------------------------ */
/* 3) gather sorted rows, fp32 -> bf16 (pad rows = 0)                  */
/* ------------------------------------------------------------------ */
__global__ void __launch_bounds__(256)
moe_gather(const float* __restrict__ x2d, const int* __restrict__ perm,
           const int* __restrict__ meta, __bf16* __restrict__ xb)
{
    const int r = blockIdx.x;
    if (r >= meta[1]) return;
    const int src = perm[r];
    __bf16* dst = xb + (size_t)r * HID + threadIdx.x * 4;
    if (src < 0) {
        dst[0] = (__bf16)0.0f; dst[1] = (__bf16)0.0f;
        dst[2] = (__bf16)0.0f; dst[3] = (__bf16)0.0f;
    } else {
        const int tok = src >> 1;            // TOPK == 2
        const float4 v = ((const float4*)(x2d + (size_t)tok * HID))[threadIdx.x];
        dst[0] = (__bf16)v.x; dst[1] = (__bf16)v.y;
        dst[2] = (__bf16)v.z; dst[3] = (__bf16)v.w;
    }
}

/* ------------------------------------------------------------------ */
/* 4) GEMM1: x[64xD] * (wi0,wi1)[DxF] -> h = silu(g0)*g1   (bf16 out)  */
/*    block = 8 waves; block tile 64 x 256; wave tile 64 x 32 x 2 mats */
/* ------------------------------------------------------------------ */
__global__ void __launch_bounds__(256)
moe_gemm1(const __bf16* __restrict__ xb, const __bf16* __restrict__ w0t,
          const __bf16* __restrict__ w1t, __bf16* __restrict__ hb,
          const int* __restrict__ meta)
{
    const int tile = blockIdx.x;
    if (tile >= meta[0]) return;
    const int e     = meta[16 + tile];
    const int rbase = meta[96 + tile];
    const int lane  = threadIdx.x & 31;
    const int wave  = threadIdx.x >> 5;
    const int nwave = blockIdx.y * 256 + wave * 32;
    const int rlo   = lane & 15;
    const int half  = lane >> 4;

    const size_t wbase = (size_t)e * FFN * HID;
    const __bf16* wg = w0t + wbase;          // [F][D] transposed
    const __bf16* wu = w1t + wbase;

    v8f acc0[4][2] = {};
    v8f acc1[4][2] = {};

    for (int k0 = 0; k0 < HID; k0 += 32) {
        v16bf af[4];
        #pragma unroll
        for (int m = 0; m < 4; ++m)
            af[m] = load_a_frag(xb + (size_t)(rbase + m * 16 + rlo) * HID
                                   + k0 + half * 16);
        #pragma unroll
        for (int n = 0; n < 2; ++n) {
            const int col = nwave + n * 16 + rlo;
            const v16bf b0 = *(const v16bf*)(wg + (size_t)col * HID + k0 + half * 16);
            const v16bf b1 = *(const v16bf*)(wu + (size_t)col * HID + k0 + half * 16);
            #pragma unroll
            for (int m = 0; m < 4; ++m) {
                acc0[m][n] = wmma_bf16(af[m], b0, acc0[m][n]);
                acc1[m][n] = wmma_bf16(af[m], b1, acc1[m][n]);
            }
        }
    }
    // epilogue: h = silu(g0) * g1  (C/D layout: vgpr j -> row half*8+j)
    #pragma unroll
    for (int m = 0; m < 4; ++m) {
        const int r = rbase + m * 16 + half * 8;
        #pragma unroll
        for (int n = 0; n < 2; ++n) {
            const int col = nwave + n * 16 + rlo;
            #pragma unroll
            for (int j = 0; j < 8; ++j) {
                const float hv = silu_mul(acc0[m][n][j], acc1[m][n][j]);
                hb[(size_t)(r + j) * FFN + col] = (__bf16)hv;
            }
        }
    }
}

/* ------------------------------------------------------------------ */
/* 5) GEMM2: h[64xF] * wo[FxD] -> y (fp32)                             */
/* ------------------------------------------------------------------ */
__global__ void __launch_bounds__(256)
moe_gemm2(const __bf16* __restrict__ hb, const __bf16* __restrict__ wot,
          float* __restrict__ yb, const int* __restrict__ meta)
{
    const int tile = blockIdx.x;
    if (tile >= meta[0]) return;
    const int e     = meta[16 + tile];
    const int rbase = meta[96 + tile];
    const int lane  = threadIdx.x & 31;
    const int wave  = threadIdx.x >> 5;
    const int nwave = blockIdx.y * 256 + wave * 32;
    const int rlo   = lane & 15;
    const int half  = lane >> 4;

    const __bf16* wd = wot + (size_t)e * HID * FFN;   // [D][F] transposed

    v8f acc[4][2] = {};

    for (int k0 = 0; k0 < FFN; k0 += 32) {
        v16bf af[4];
        #pragma unroll
        for (int m = 0; m < 4; ++m)
            af[m] = load_a_frag(hb + (size_t)(rbase + m * 16 + rlo) * FFN
                                   + k0 + half * 16);
        #pragma unroll
        for (int n = 0; n < 2; ++n) {
            const int col = nwave + n * 16 + rlo;
            const v16bf b = *(const v16bf*)(wd + (size_t)col * FFN + k0 + half * 16);
            #pragma unroll
            for (int m = 0; m < 4; ++m)
                acc[m][n] = wmma_bf16(af[m], b, acc[m][n]);
        }
    }
    #pragma unroll
    for (int m = 0; m < 4; ++m) {
        const int r = rbase + m * 16 + half * 8;
        #pragma unroll
        for (int n = 0; n < 2; ++n) {
            const int col = nwave + n * 16 + rlo;
            #pragma unroll
            for (int j = 0; j < 8; ++j)
                yb[(size_t)(r + j) * HID + col] = acc[m][n][j];
        }
    }
}

/* ------------------------------------------------------------------ */
/* 6) unpermute + routing-weight combine (fp32)                        */
/* ------------------------------------------------------------------ */
__global__ void __launch_bounds__(256)
moe_combine(const float* __restrict__ yb, const float* __restrict__ tw,
            const int* __restrict__ inv, float* __restrict__ out)
{
    const int idx = blockIdx.x * 256 + threadIdx.x;   // over TOKENS * HID/4
    if (idx >= TOKENS * (HID / 4)) return;
    const int t  = idx / (HID / 4);
    const int dq = idx % (HID / 4);
    const int r0 = inv[t * 2 + 0];
    const int r1 = inv[t * 2 + 1];
    const float w0 = tw[t * 2 + 0];
    const float w1 = tw[t * 2 + 1];
    const float4 a = ((const float4*)(yb + (size_t)r0 * HID))[dq];
    const float4 b = ((const float4*)(yb + (size_t)r1 * HID))[dq];
    float4 o;
    o.x = w0 * a.x + w1 * b.x;
    o.y = w0 * a.y + w1 * b.y;
    o.z = w0 * a.z + w1 * b.z;
    o.w = w0 * a.w + w1 * b.w;
    ((float4*)out)[idx] = o;
}

/* ------------------------------------------------------------------ */
/* launch                                                              */
/* ------------------------------------------------------------------ */
extern "C" void kernel_launch(void* const* d_in, const int* in_sizes, int n_in,
                              void* d_out, int out_size, void* d_ws, size_t ws_size,
                              hipStream_t stream)
{
    const float* x2d = (const float*)d_in[0];   // [T, D] fp32
    const float* tw  = (const float*)d_in[1];   // [T, K] fp32
    const int*   ids = (const int*)  d_in[2];   // [T, K] int32
    const float* wi0 = (const float*)d_in[3];   // [E, D, F]
    const float* wi1 = (const float*)d_in[4];   // [E, D, F]
    const float* wo  = (const float*)d_in[5];   // [E, F, D]
    float* out = (float*)d_out;
    (void)in_sizes; (void)n_in; (void)out_size; (void)ws_size;

    // workspace carve-out (256B aligned)
    char* p = (char*)d_ws;
    auto carve = [&](size_t bytes) {
        void* r = (void*)p;
        p += (bytes + 255) & ~(size_t)255;
        return r;
    };
    const size_t wElems = (size_t)EXPERTS * HID * FFN;
    __bf16* w0t  = (__bf16*)carve(wElems * sizeof(__bf16));      // [E][F][D]
    __bf16* w1t  = (__bf16*)carve(wElems * sizeof(__bf16));      // [E][F][D]
    __bf16* wot  = (__bf16*)carve(wElems * sizeof(__bf16));      // [E][D][F]
    __bf16* xb   = (__bf16*)carve((size_t)MAX_PROWS * HID * sizeof(__bf16));
    __bf16* hb   = (__bf16*)carve((size_t)MAX_PROWS * FFN * sizeof(__bf16));
    float*  yb   = (float*) carve((size_t)MAX_PROWS * HID * sizeof(float));
    int*    meta = (int*)   carve(1024 * sizeof(int));
    int*    perm = (int*)   carve(MAX_PROWS * sizeof(int));
    int*    inv  = (int*)   carve(NROWS * sizeof(int));

    const dim3 tblk(32, 8, 1);
    // wi: [E][D][F] -> [E][F][D]
    cvt_transpose<<<dim3(FFN / 32, HID / 32, EXPERTS), tblk, 0, stream>>>(wi0, w0t, HID, FFN);
    cvt_transpose<<<dim3(FFN / 32, HID / 32, EXPERTS), tblk, 0, stream>>>(wi1, w1t, HID, FFN);
    // wo: [E][F][D] -> [E][D][F]
    cvt_transpose<<<dim3(HID / 32, FFN / 32, EXPERTS), tblk, 0, stream>>>(wo, wot, FFN, HID);

    moe_prep<<<1, 256, 0, stream>>>(ids, meta, perm, inv);
    moe_gather<<<MAX_PROWS, 256, 0, stream>>>(x2d, perm, meta, xb);

    moe_gemm1<<<dim3(MAX_TILES, FFN / 256, 1), 256, 0, stream>>>(xb, w0t, w1t, hb, meta);
    moe_gemm2<<<dim3(MAX_TILES, HID / 256, 1), 256, 0, stream>>>(hb, wot, yb, meta);

    moe_combine<<<(TOKENS * (HID / 4) + 255) / 256, 256, 0, stream>>>(yb, tw, inv, out);
}
